// CausalSelfAttention_56556129354396
// MI455X (gfx1250) — matrix operand order repaired and checked by reading, not compiled, not run on hardware
//
#include <hip/hip_runtime.h>

// ---------------------------------------------------------------------------
// Causal self-attention forward for MI455X (gfx1250, wave32, WMMA bf16).
// Pipeline: cast/transpose -> QKV GEMM (async global->LDS staging, wmma bf16,
// f32 accum, bias+scale, head-major scatter) -> flash attention (online
// softmax, wmma) -> proj GEMM.
// ---------------------------------------------------------------------------

typedef __bf16 bf16t;
typedef __attribute__((ext_vector_type(16))) __bf16 v16bf;
typedef __attribute__((ext_vector_type(8)))  __bf16 v8bf;
typedef __attribute__((ext_vector_type(8)))  float  v8f;

#define EMBED   1024
#define NHEAD   16
#define HDIM    64
#define TSEQ    2048
#define NBATCH  2
#define MTOK    (NBATCH*TSEQ)   // 4096 tokens
#define N_QKV   (3*EMBED)       // 3072

#define BM 128
#define BN 128
#define BK 32
#define LK (BK+8)               // LDS row stride (pad: 80B -> conflict-free b128)

// ---- CDNA5 async global->LDS copy (ASYNCcnt-tracked DMA, 16B per lane) ----

static __device__ __forceinline__ void async_copy_b128(void* lds, const void* gptr){
  // LDS byte address = low 32 bits of the generic pointer (addr[31:0]).
  unsigned ldsa = (unsigned)(unsigned long long)(uintptr_t)lds;
  unsigned long long ga = (unsigned long long)(uintptr_t)gptr;
  asm volatile("global_load_async_to_lds_b128 %0, %1, off"
               :: "v"(ldsa), "v"(ga)
               : "memory");
}

static __device__ __forceinline__ void wait_async0(){
  asm volatile("s_wait_asynccnt 0x0" ::: "memory");
}

// ---- fragment helpers (contiguous k order in A and B, 16-bit 16x16x32) ------

static __device__ __forceinline__ v16bf join8(v8bf lo, v8bf hi){
  v16bf r;
#pragma unroll
  for (int i = 0; i < 8; ++i){ r[i] = lo[i]; r[i+8] = hi[i]; }
  return r;
}

// A-matrix 16x32 bf16: lane<16 -> M=lane, K=0..15;
//                      lane>=16 -> M=lane-16, K=16..31.
template<int STRIDE>
static __device__ __forceinline__ v16bf load_afrag(const bf16t* base){
  const int lane = threadIdx.x & 31;
  const bf16t* p = base + (lane & 15) * STRIDE + ((lane < 16) ? 0 : 16);
  return join8(*(const v8bf*)p, *(const v8bf*)(p + 8));
}

// B-matrix 32x16 bf16 read from an N-major buffer (row n contiguous over k):
// lane<16 -> N=lane, K=0..15; lane>=16 -> N=lane-16, K=16..31 (32B contiguous).
template<int STRIDE>
static __device__ __forceinline__ v16bf load_bfrag(const bf16t* base){
  const int lane = threadIdx.x & 31;
  const bf16t* p = base + (lane & 15) * STRIDE + ((lane < 16) ? 0 : 16);
  return join8(*(const v8bf*)p, *(const v8bf*)(p + 8));
}

static __device__ __forceinline__ v8f wmma_bf16(v16bf a, v16bf b, v8f c){
  return __builtin_amdgcn_wmma_f32_16x16x32_bf16(
      /*neg_a=*/false, a, /*neg_b=*/false, b,
      /*c_mod=*/(short)0, c, /*reuse_a=*/false, /*reuse_b=*/false);
}

// ---- conversion kernels ---------------------------------------------------

__global__ void cvt_cast_kernel(const float* __restrict__ in,
                                bf16t* __restrict__ out, int n){
  int i = blockIdx.x * blockDim.x + threadIdx.x;
  if (i < n) out[i] = (bf16t)in[i];
}

// in: [K][N] row-major f32 -> out: [N][K] bf16 (N-major for B-fragments)
__global__ void cvt_transpose_kernel(const float* __restrict__ in,
                                     bf16t* __restrict__ out, int K, int N){
  int i = blockIdx.x * blockDim.x + threadIdx.x;
  if (i < K * N){
    int k = i / N, n = i - k * N;
    out[(size_t)n * K + k] = (bf16t)in[i];
  }
}

// ---- tiled WMMA GEMM: C = A[M,K] * BT[N,K]^T (+bias), two epilogues -------
// mode 0: QKV -> scatter q (scaled 0.125) / k to [B,H,T,D], v to [B,H,D,T]
// mode 1: proj -> fp32 out[M,EMBED]

__global__ __launch_bounds__(256) void gemm_qkv_proj(
    const bf16t* __restrict__ A, const bf16t* __restrict__ BT,
    const float* __restrict__ bias, int Kdim, int mode,
    bf16t* __restrict__ qO, bf16t* __restrict__ kO, bf16t* __restrict__ vTO,
    float* __restrict__ out)
{
  __shared__ bf16t As[2][BM][LK];
  __shared__ bf16t Bs[2][BN][LK];
  const int tid  = threadIdx.x;
  const int lane = tid & 31;
  const int wave = tid >> 5;       // 8 waves
  const int wrow = wave >> 1;      // 0..3 -> 32-row slice
  const int wcol = wave & 1;       // 0..1 -> 64-col slice
  const int m0 = blockIdx.y * BM;
  const int n0 = blockIdx.x * BN;

  v8f acc[2][4];
#pragma unroll
  for (int mt = 0; mt < 2; ++mt)
#pragma unroll
    for (int nt = 0; nt < 4; ++nt)
#pragma unroll
      for (int i = 0; i < 8; ++i) acc[mt][nt][i] = 0.0f;

  const int srow = tid >> 1;            // 0..127
  const int sch  = (tid & 1) * 16;      // 0 or 16
  const bf16t* gA = A  + (size_t)(m0 + srow) * Kdim + sch;
  const bf16t* gB = BT + (size_t)(n0 + srow) * Kdim + sch;

  const int KT = Kdim / BK;
  // prologue: async DMA stage into buffer 0
  async_copy_b128(&As[0][srow][sch],     gA);
  async_copy_b128(&As[0][srow][sch + 8], gA + 8);
  async_copy_b128(&Bs[0][srow][sch],     gB);
  async_copy_b128(&Bs[0][srow][sch + 8], gB + 8);
  wait_async0();
  __syncthreads();

  for (int kt = 0; kt < KT; ++kt){
    const int buf = kt & 1;
    if (kt + 1 < KT){
      const bf16t* a = gA + (size_t)(kt + 1) * BK;
      const bf16t* b = gB + (size_t)(kt + 1) * BK;
      if (kt + 2 < KT){                       // global_prefetch_b8 two tiles out
        __builtin_prefetch(a + BK, 0, 1);
        __builtin_prefetch(b + BK, 0, 1);
      }
      // async DMA stage of next K-tile overlaps with WMMA on current buffer
      async_copy_b128(&As[buf ^ 1][srow][sch],     a);
      async_copy_b128(&As[buf ^ 1][srow][sch + 8], a + 8);
      async_copy_b128(&Bs[buf ^ 1][srow][sch],     b);
      async_copy_b128(&Bs[buf ^ 1][srow][sch + 8], b + 8);
    }
    v16bf af[2], bfv[4];
#pragma unroll
    for (int mt = 0; mt < 2; ++mt)
      af[mt] = load_afrag<LK>(&As[buf][wrow * 32 + mt * 16][0]);
#pragma unroll
    for (int nt = 0; nt < 4; ++nt)
      bfv[nt] = load_bfrag<LK>(&Bs[buf][wcol * 64 + nt * 16][0]);
#pragma unroll
    for (int mt = 0; mt < 2; ++mt)
#pragma unroll
      for (int nt = 0; nt < 4; ++nt)
        acc[mt][nt] = wmma_bf16(af[mt], bfv[nt], acc[mt][nt]);
    wait_async0();                // own async stage complete before barrier
    __syncthreads();
  }

  // epilogue: C layout = lane<16: (M=r, N=lane); lane>=16: (M=r+8, N=lane-16)
#pragma unroll
  for (int mt = 0; mt < 2; ++mt){
#pragma unroll
    for (int nt = 0; nt < 4; ++nt){
      const int n = n0 + wcol * 64 + nt * 16 + (lane & 15);
      const float bv = bias[n];
#pragma unroll
      for (int r = 0; r < 8; ++r){
        const int m = m0 + wrow * 32 + mt * 16 + r + ((lane < 16) ? 0 : 8);
        const float v = acc[mt][nt][r] + bv;
        if (mode == 0){
          const int bb = m >> 11;          // batch
          const int t  = m & (TSEQ - 1);   // token
          if (n < EMBED){
            const int h = n >> 6, d = n & 63;
            qO[(((size_t)(bb * NHEAD + h)) * TSEQ + t) * HDIM + d] =
                (bf16t)(v * 0.125f);       // fold 1/sqrt(64)
          } else if (n < 2 * EMBED){
            const int f = n - EMBED;
            const int h = f >> 6, d = f & 63;
            kO[(((size_t)(bb * NHEAD + h)) * TSEQ + t) * HDIM + d] = (bf16t)v;
          } else {
            const int f = n - 2 * EMBED;
            const int h = f >> 6, d = f & 63;
            vTO[(((size_t)(bb * NHEAD + h)) * HDIM + d) * TSEQ + t] = (bf16t)v;
          }
        } else {
          out[(size_t)m * EMBED + n] = v;
        }
      }
    }
  }
}

// ---- flash attention: 8 waves/block, 16 query rows per wave ----------------

__global__ __launch_bounds__(256) void attn_kernel(
    const bf16t* __restrict__ Q, const bf16t* __restrict__ K,
    const bf16t* __restrict__ VT, bf16t* __restrict__ Y)
{
  __shared__ bf16t Ps[8][16][LK];   // wave-private P staging (C-layout -> A-frag)
  const int lane = threadIdx.x & 31;
  const int wave = threadIdx.x >> 5;
  const int bh = blockIdx.y;
  const int b  = bh >> 4;
  const int h  = bh & 15;
  const bf16t* q  = Q  + (size_t)bh * TSEQ * HDIM;
  const bf16t* kk = K  + (size_t)bh * TSEQ * HDIM;
  const bf16t* vt = VT + (size_t)bh * HDIM * TSEQ;
  const int qBase = blockIdx.x * 128 + wave * 16;

  v16bf qf[2];
#pragma unroll
  for (int dB = 0; dB < 2; ++dB)
    qf[dB] = load_afrag<HDIM>(q + (size_t)qBase * HDIM + dB * 32);

  float mrow[8], lrow[8];
  v8f yacc[4];
#pragma unroll
  for (int r = 0; r < 8; ++r){ mrow[r] = -1e30f; lrow[r] = 0.0f; }
#pragma unroll
  for (int nt = 0; nt < 4; ++nt)
#pragma unroll
    for (int i = 0; i < 8; ++i) yacc[nt][i] = 0.0f;

  const int kEnd = qBase + 16;      // causal: keys <= max row in this wave
  for (int kb = 0; kb < kEnd; kb += 32){
    v8f s0, s1;
#pragma unroll
    for (int i = 0; i < 8; ++i){ s0[i] = 0.0f; s1[i] = 0.0f; }
#pragma unroll
    for (int dB = 0; dB < 2; ++dB){
      v16bf kf0 = load_bfrag<HDIM>(kk + (size_t)kb * HDIM + dB * 32);
      v16bf kf1 = load_bfrag<HDIM>(kk + (size_t)(kb + 16) * HDIM + dB * 32);
      s0 = wmma_bf16(qf[dB], kf0, s0);
      s1 = wmma_bf16(qf[dB], kf1, s1);
    }
    if (kb + 31 > qBase){           // diagonal block(s): causal mask
      const int rbase = qBase + ((lane < 16) ? 0 : 8);
      const int c0 = kb + (lane & 15);
      const int c1 = c0 + 16;
#pragma unroll
      for (int r = 0; r < 8; ++r){
        const int m = rbase + r;
        s0[r] = (c0 <= m) ? s0[r] : -1e30f;
        s1[r] = (c1 <= m) ? s1[r] : -1e30f;
      }
    }
    // online softmax: a full row lives in one 16-lane half -> xor reductions
#pragma unroll
    for (int r = 0; r < 8; ++r){
      float mx = fmaxf(s0[r], s1[r]);
#pragma unroll
      for (int off = 1; off < 16; off <<= 1)
        mx = fmaxf(mx, __shfl_xor(mx, off, 32));
      const float mnew = fmaxf(mrow[r], mx);
      const float rescale = __expf(mrow[r] - mnew);
      mrow[r] = mnew;
      const float p0 = __expf(s0[r] - mnew);
      const float p1 = __expf(s1[r] - mnew);
      float rs = p0 + p1;
#pragma unroll
      for (int off = 1; off < 16; off <<= 1)
        rs += __shfl_xor(rs, off, 32);
      lrow[r] = lrow[r] * rescale + rs;
#pragma unroll
      for (int nt = 0; nt < 4; ++nt) yacc[nt][r] *= rescale;
      s0[r] = p0; s1[r] = p1;
    }
    // C-layout -> LDS -> A-fragment (wave-private; DS ops are in-order)
    const int prow = (lane < 16) ? 0 : 8;
    const int pcol = lane & 15;
#pragma unroll
    for (int r = 0; r < 8; ++r){
      Ps[wave][prow + r][pcol]      = (bf16t)s0[r];
      Ps[wave][prow + r][pcol + 16] = (bf16t)s1[r];
    }
    v16bf pf = load_afrag<LK>(&Ps[wave][0][0]);
#pragma unroll
    for (int nt = 0; nt < 4; ++nt){
      v16bf vf = load_bfrag<TSEQ>(vt + (size_t)(nt * 16) * TSEQ + kb);
      yacc[nt] = wmma_bf16(pf, vf, yacc[nt]);
    }
  }

#pragma unroll
  for (int r = 0; r < 8; ++r){
    const float inv = 1.0f / lrow[r];
    const int t = qBase + r + ((lane < 16) ? 0 : 8);
    const size_t base = ((size_t)(b * TSEQ + t)) * EMBED + h * HDIM;
#pragma unroll
    for (int nt = 0; nt < 4; ++nt){
      const int d = nt * 16 + (lane & 15);
      Y[base + d] = (bf16t)(yacc[nt][r] * inv);
    }
  }
}

// ---- launch ---------------------------------------------------------------

extern "C" void kernel_launch(void* const* d_in, const int* in_sizes, int n_in,
                              void* d_out, int out_size, void* d_ws, size_t ws_size,
                              hipStream_t stream)
{
  const float* x      = (const float*)d_in[0];
  const float* w_attn = (const float*)d_in[1];
  const float* b_attn = (const float*)d_in[2];
  const float* w_proj = (const float*)d_in[3];
  const float* b_proj = (const float*)d_in[4];
  float* out = (float*)d_out;
  (void)in_sizes; (void)n_in; (void)out_size; (void)ws_size;

  char* ws = (char*)d_ws;
  size_t off = 0;
  auto alloc = [&](size_t bytes) -> char* {
    char* p = ws + off;
    off += (bytes + 255) & ~(size_t)255;
    return p;
  };
  // ~50.3 MB total workspace
  bf16t* xb  = (bf16t*)alloc((size_t)MTOK * EMBED * sizeof(bf16t));
  bf16t* waT = (bf16t*)alloc((size_t)N_QKV * EMBED * sizeof(bf16t));
  bf16t* wpT = (bf16t*)alloc((size_t)EMBED * EMBED * sizeof(bf16t));
  bf16t* qB  = (bf16t*)alloc((size_t)MTOK * EMBED * sizeof(bf16t));
  bf16t* kB  = (bf16t*)alloc((size_t)MTOK * EMBED * sizeof(bf16t));
  bf16t* vTB = (bf16t*)alloc((size_t)MTOK * EMBED * sizeof(bf16t));
  bf16t* yB  = (bf16t*)alloc((size_t)MTOK * EMBED * sizeof(bf16t));

  {
    int n = MTOK * EMBED;
    cvt_cast_kernel<<<n / 256, 256, 0, stream>>>(x, xb, n);
  }
  {
    int n = EMBED * N_QKV;
    cvt_transpose_kernel<<<n / 256, 256, 0, stream>>>(w_attn, waT, EMBED, N_QKV);
  }
  {
    int n = EMBED * EMBED;
    cvt_transpose_kernel<<<n / 256, 256, 0, stream>>>(w_proj, wpT, EMBED, EMBED);
  }
  {
    dim3 grid(N_QKV / BN, MTOK / BM);   // (24, 32)
    gemm_qkv_proj<<<grid, 256, 0, stream>>>(xb, waT, b_attn, EMBED, 0,
                                            qB, kB, vTB, nullptr);
  }
  {
    dim3 grid(TSEQ / 128, NBATCH * NHEAD); // (16, 32)
    attn_kernel<<<grid, 256, 0, stream>>>(qB, kB, vTB, yB);
  }
  {
    dim3 grid(EMBED / BN, MTOK / BM);   // (8, 32)
    gemm_qkv_proj<<<grid, 256, 0, stream>>>(yB, wpT, b_proj, EMBED, 1,
                                            nullptr, nullptr, nullptr, out);
  }
}
